// MultiHeadAttention_33088428048805
// MI455X (gfx1250) — matrix operand order repaired and checked by reading, not compiled, not run on hardware
//
#include <hip/hip_runtime.h>
#include <hip/hip_bf16.h>
#include <math.h>

// ---------------------------------------------------------------------------
// MultiHeadAttention for MI455X (gfx1250, wave32, WMMA bf16 + TDM staging)
//   q = x@Wq^T+bq ; k,v likewise ; flash attention (causal) ; out = a@Wo^T+bo
// ---------------------------------------------------------------------------

#define D_MODEL 1024
#define N_HEADS 16
#define D_HEAD  64
#define BATCH   4
#define SEQ     2048

typedef __attribute__((ext_vector_type(16))) __bf16        bf16x16;
typedef __attribute__((ext_vector_type(8)))  float         f32x8;
typedef __attribute__((ext_vector_type(4)))  unsigned int  u32x4;
typedef __attribute__((ext_vector_type(8)))  int           i32x8;
typedef __attribute__((ext_vector_type(4)))  int           i32x4;

#if defined(__gfx1250__) && __has_builtin(__builtin_amdgcn_tensor_load_to_lds) && \
    __has_builtin(__builtin_amdgcn_s_wait_tensorcnt)
#define USE_TDM 1
#else
#define USE_TDM 0
#endif

union FragBF {
    bf16x16 v;
    uint4   u4[2];
    __bf16  h[16];
};

static __device__ inline f32x8 zero8() {
    f32x8 z;
#pragma unroll
    for (int i = 0; i < 8; ++i) z[i] = 0.0f;
    return z;
}

// B-fragment (32x16, K x N, 16-bit): lane n holds 16 contiguous K values at
// K offset 16*(lane/16).  Source must be reduction-contiguous.
static __device__ inline bf16x16 load_frag_contig(const __bf16* p) {
    FragBF f;
    f.u4[0] = *reinterpret_cast<const uint4*>(p);
    f.u4[1] = *reinterpret_cast<const uint4*>(p + 8);
    return f.v;
}

// A-fragment (16x32, M x K, 16-bit): lane holds two 8-elem chunks of its row
// at K offsets 16*half and 16*half+8.
static __device__ inline bf16x16 load_frag_a_bf16(const __bf16* row, int k0, int k1) {
    FragBF f;
    f.u4[0] = *reinterpret_cast<const uint4*>(row + k0);
    f.u4[1] = *reinterpret_cast<const uint4*>(row + k1);
    return f.v;
}

static __device__ inline bf16x16 load_frag_a_f32(const float* row, int k0, int k1) {
    FragBF f;
#pragma unroll
    for (int i = 0; i < 8; ++i) f.h[i]     = (__bf16)row[k0 + i];
#pragma unroll
    for (int i = 0; i < 8; ++i) f.h[8 + i] = (__bf16)row[k1 + i];
    return f.v;
}

static __device__ inline f32x8 wmma_bf16(bf16x16 a, bf16x16 b, f32x8 c) {
    // D = A(16x32) * B(32x16) + C
    return __builtin_amdgcn_wmma_f32_16x16x32_bf16(false, a, false, b, (short)0, c,
                                                   false, false);
}

#if USE_TDM
// ---------------------------------------------------------------------------
// Issue a TDM load of a 64(n) x 32(k) bf16 tile of W[N,K] into LDS.
// D# built per CDNA5 ISA sec 8.3/8.4 (group0 128b, group1 256b; groups 2/3
// zero => 2D tile, VADDR2/3 semantics unused since tile_dim2/3 = 0).
// ---------------------------------------------------------------------------
static __device__ inline void tdm_stage_b_tile(const __bf16* tile_gaddr,
                                               unsigned int lds_byte_off,
                                               int K, int N) {
    const unsigned long long ga = (unsigned long long)(uintptr_t)tile_gaddr;
    u32x4 g0;
    g0[0] = 1u;                                       // count=1, is_restore=0
    g0[1] = lds_byte_off;                             // lds_addr[31:0]
    g0[2] = (unsigned int)(ga & 0xffffffffu);         // global_addr[31:0]
    g0[3] = (unsigned int)((ga >> 32) & 0x01ffffffu)  // global_addr[56:32]
            | (2u << 30);                             // type=2 ("image")
    i32x8 g1;
    g1[0] = 0x00010000;                               // data_size=1 -> 2 bytes
    g1[1] = (int)(((unsigned)K & 0xffffu) << 16);     // tensor_dim0[15:0]
    g1[2] = (int)((((unsigned)K >> 16) & 0xffffu) |   // tensor_dim0[31:16]
                  (((unsigned)N & 0xffffu) << 16));   // tensor_dim1[15:0]
    g1[3] = (int)((((unsigned)N >> 16) & 0xffffu) |   // tensor_dim1[31:16]
                  (32u << 16));                       // tile_dim0 = 32 (k)
    g1[4] = 64;                                       // tile_dim1 = 64 (n), tile_dim2=0
    g1[5] = K;                                        // tensor_dim0_stride[31:0]
    g1[6] = 0;                                        // stride hi / dim1_stride lo
    g1[7] = 0;
    i32x4 z4 = {0, 0, 0, 0};
#if __clang_major__ >= 23
    i32x8 z8 = {0, 0, 0, 0, 0, 0, 0, 0};
    __builtin_amdgcn_tensor_load_to_lds(g0, g1, z4, z4, z8, 0);
#else
    __builtin_amdgcn_tensor_load_to_lds(g0, g1, z4, z4, 0);
#endif
}
#endif  // USE_TDM

// ---------------------------------------------------------------------------
// fp32 -> bf16 conversion (weights)
// ---------------------------------------------------------------------------
__global__ void cvt_f32_bf16_kernel(const float* __restrict__ in,
                                    __bf16* __restrict__ out, int n) {
    int i = blockIdx.x * blockDim.x + threadIdx.x;
    if (i < n) out[i] = (__bf16)in[i];
}

// ---------------------------------------------------------------------------
// GEMM: O[M,N] = A[M,K] * W[N,K]^T + bias[N]
//   A_BF16:     A is bf16 (else fp32, converted in-register)
//   STORE_MODE: 0 = fp32 row-major, 1 = bf16 row-major,
//               2 = bf16 V-transpose store into [B][H][D_HEAD][SEQ]
// Block: 128 threads (4 waves); wave w owns a 16x64 output tile.
// Weight tile (64 x 32 bf16, 4KB) staged once per WG per K-step via TDM,
// double-buffered; waves read B-fragments from LDS (ds_load_b128).
// ---------------------------------------------------------------------------
template <bool A_BF16, int STORE_MODE>
__global__ __launch_bounds__(128)
void gemm_wmma_kernel(const void* __restrict__ Av,
                      const __bf16* __restrict__ Bw,
                      const float* __restrict__ bias,
                      void* __restrict__ Ov,
                      int M, int N, int K) {
    __shared__ __align__(16) __bf16 btile[2][64][32];  // [buf][n_local][k_local]

    const int lane = threadIdx.x & 31;
    const int wave = threadIdx.x >> 5;
    const int ln   = lane & 15;
    const int half = lane >> 4;

    const int m0 = blockIdx.y * 64 + wave * 16;
    const int n0 = blockIdx.x * 64;
    const int arow = m0 + ln;

    f32x8 acc[4];
#pragma unroll
    for (int t = 0; t < 4; ++t) acc[t] = zero8();

    const float*  Af = (const float*)Av;
    const __bf16* Ab = (const __bf16*)Av;

#if USE_TDM
    // Prologue: stage tile for k=0 into buffer 0 (one DMA per workgroup).
    if (wave == 0) {
        tdm_stage_b_tile(Bw + (size_t)n0 * K,
                         (unsigned int)(uintptr_t)&btile[0][0][0], K, N);
    }
#endif

    const int nsteps = K / 32;
    for (int kstep = 0; kstep < nsteps; ++kstep) {
        const int k   = kstep * 32;
        const int cur = kstep & 1;

#if USE_TDM
        if (wave == 0) {
            if (kstep + 1 < nsteps) {
                // Stage next tile into the other buffer, then wait for the
                // oldest (current) DMA only: TENSORcnt is in-order per wave.
                tdm_stage_b_tile(Bw + (size_t)n0 * K + (k + 32),
                                 (unsigned int)(uintptr_t)&btile[cur ^ 1][0][0], K, N);
                __builtin_amdgcn_s_wait_tensorcnt(1);
            } else {
                __builtin_amdgcn_s_wait_tensorcnt(0);
            }
        }
        __syncthreads();  // publish LDS tile to all 4 waves
#else
        // Fallback: cooperative synchronous stage (single buffer semantics).
        __syncthreads();  // previous reads done before overwrite
        {
            uint4*       dst = (uint4*)&btile[cur][0][0];
            const __bf16* srcbase = Bw + (size_t)n0 * K + k;
#pragma unroll
            for (int u = threadIdx.x; u < 256; u += 128) {  // 256 uint4 = 4KB
                const int rown  = u >> 2;
                const int chunk = u & 3;
                dst[u] = *reinterpret_cast<const uint4*>(srcbase +
                                                         (size_t)rown * K + chunk * 8);
            }
        }
        __syncthreads();
#endif

        const int k0 = k + 16 * half;
        const int k1 = k + 16 * half + 8;
        bf16x16 afrag;
        if (A_BF16) {
            const __bf16* ap = Ab + (size_t)arow * K;
            if (k + 32 < K) __builtin_prefetch(ap + k + 32, 0, 0);
            afrag = load_frag_a_bf16(ap, k0, k1);
        } else {
            const float* ap = Af + (size_t)arow * K;
            if (k + 32 < K) __builtin_prefetch(ap + k + 32, 0, 0);
            afrag = load_frag_a_f32(ap, k0, k1);
        }

#pragma unroll
        for (int t = 0; t < 4; ++t) {
            // B-fragment from LDS: lane n = 16t+ln, 16 contiguous k at 16*half.
            FragBF bf;
            const __bf16* bp = &btile[cur][16 * t + ln][16 * half];
            bf.u4[0] = *reinterpret_cast<const uint4*>(bp);
            bf.u4[1] = *reinterpret_cast<const uint4*>(bp + 8);
            acc[t] = wmma_bf16(afrag, bf.v, acc[t]);
        }

        __syncthreads();  // all waves done with btile[cur] before it is re-staged
    }

#pragma unroll
    for (int t = 0; t < 4; ++t) {
        const int n = n0 + 16 * t + ln;
        const float bv = bias[n];
#pragma unroll
        for (int r = 0; r < 8; ++r) {
            const int row = m0 + r + 8 * half;  // C layout: VGPR r -> M=r (+8 hi half)
            const float val = acc[t][r] + bv;
            if (STORE_MODE == 0) {
                ((float*)Ov)[(size_t)row * N + n] = val;
            } else if (STORE_MODE == 1) {
                ((__bf16*)Ov)[(size_t)row * N + n] = (__bf16)val;
            } else {
                const int b = row / SEQ, s = row % SEQ;
                const int h = n / D_HEAD, d = n % D_HEAD;
                ((__bf16*)Ov)[((size_t)(b * N_HEADS + h) * D_HEAD + d) * SEQ + s] =
                    (__bf16)val;
            }
        }
    }
}

// ---------------------------------------------------------------------------
// Causal flash attention. One wave = 16 query rows of one (b,h).
//   Qp, Kp : bf16 [B*SEQ, D_MODEL]  (head h = columns h*64..h*64+63)
//   Vt     : bf16 [B, H, D_HEAD, SEQ]  (pre-transposed V)
//   Out    : bf16 [B*SEQ, D_MODEL]
// Block: 128 threads (4 independent waves -> 64 query rows per WG).
// ---------------------------------------------------------------------------
__global__ __launch_bounds__(128)
void attention_kernel(const __bf16* __restrict__ Qp,
                      const __bf16* __restrict__ Kp,
                      const __bf16* __restrict__ Vt,
                      __bf16* __restrict__ Out) {
    __shared__ __align__(16) __bf16 plds[4][16][32];  // per-wave P tile (M x Kblk)

    const int lane = threadIdx.x & 31;
    const int wave = threadIdx.x >> 5;
    const int ln   = lane & 15;
    const int half = lane >> 4;

    const int qtiles = SEQ / 64;
    const int qt = blockIdx.x % qtiles;
    const int bh = blockIdx.x / qtiles;
    const int h  = bh % N_HEADS;
    const int b  = bh / N_HEADS;

    const int qbase = qt * 64 + wave * 16;
    const int q_max = qbase + 15;

    // Preload Q A-fragments for d=[0,32) and d=[32,64)
    const __bf16* qrow =
        Qp + (size_t)(b * SEQ + qbase + ln) * D_MODEL + h * D_HEAD;
    bf16x16 qf[2];
#pragma unroll
    for (int dblk = 0; dblk < 2; ++dblk) {
        FragBF f;
        f.u4[0] = *reinterpret_cast<const uint4*>(qrow + dblk * 32 + 16 * half);
        f.u4[1] = *reinterpret_cast<const uint4*>(qrow + dblk * 32 + 16 * half + 8);
        qf[dblk] = f.v;
    }

    f32x8 acc[4];
#pragma unroll
    for (int t = 0; t < 4; ++t) acc[t] = zero8();
    float mrow[8], lrow[8];
#pragma unroll
    for (int r = 0; r < 8; ++r) { mrow[r] = -1.0e30f; lrow[r] = 0.0f; }

    const __bf16* Vhead = Vt + (size_t)(b * N_HEADS + h) * D_HEAD * SEQ;
    const float inv_sqrt_d = 0.125f;  // 1/sqrt(64)

    for (int kb = 0; kb <= q_max; kb += 32) {
        // ---- scores S[q, kb..kb+31] via 2x2 WMMAs over d ----
        float sc[2][8];
#pragma unroll
        for (int j = 0; j < 2; ++j) {
            f32x8 c = zero8();
            const __bf16* krow =
                Kp + (size_t)(b * SEQ + kb + 16 * j + ln) * D_MODEL + h * D_HEAD;
            c = wmma_bf16(qf[0], load_frag_contig(krow + 16 * half), c);
            c = wmma_bf16(qf[1], load_frag_contig(krow + 32 + 16 * half), c);
            const int keyc = kb + 16 * j + ln;
#pragma unroll
            for (int r = 0; r < 8; ++r) {
                const int q = qbase + r + 8 * half;
                sc[j][r] = (keyc > q) ? -1.0e30f : c[r] * inv_sqrt_d;
            }
        }

        // ---- online softmax (row stats replicated across 16-lane halves) ----
#pragma unroll
        for (int r = 0; r < 8; ++r) {
            float bm = fmaxf(sc[0][r], sc[1][r]);
#pragma unroll
            for (int m = 1; m < 16; m <<= 1) bm = fmaxf(bm, __shfl_xor(bm, m));
            const float mnew  = fmaxf(mrow[r], bm);
            const float scale = __expf(mrow[r] - mnew);
            const float p0 = __expf(sc[0][r] - mnew);
            const float p1 = __expf(sc[1][r] - mnew);
            float rs = p0 + p1;
#pragma unroll
            for (int m = 1; m < 16; m <<= 1) rs += __shfl_xor(rs, m);
            lrow[r] = lrow[r] * scale + rs;
            mrow[r] = mnew;
#pragma unroll
            for (int t = 0; t < 4; ++t) acc[t][r] *= scale;
            // C-layout -> row-major P tile in LDS (re-swizzle for A-fragment)
            plds[wave][r + 8 * half][ln]      = (__bf16)p0;
            plds[wave][r + 8 * half][16 + ln] = (__bf16)p1;
        }

        // ---- reload P as A-fragment (16x32) from LDS ----
        FragBF pf;
        pf.u4[0] = *reinterpret_cast<const uint4*>(&plds[wave][ln][16 * half]);
        pf.u4[1] = *reinterpret_cast<const uint4*>(&plds[wave][ln][16 * half + 8]);

        // ---- O += P(16x32) * V(32x16) per 16-wide d tile ----
#pragma unroll
        for (int t = 0; t < 4; ++t) {
            const __bf16* vrow = Vhead + (size_t)(16 * t + ln) * SEQ + kb + 16 * half;
            acc[t] = wmma_bf16(pf.v, load_frag_contig(vrow), acc[t]);
        }
    }

    // ---- epilogue: divide by row sum, store bf16 ----
#pragma unroll
    for (int t = 0; t < 4; ++t) {
#pragma unroll
        for (int r = 0; r < 8; ++r) {
            const int q = qbase + r + 8 * half;
            const float val = acc[t][r] / lrow[r];
            Out[(size_t)(b * SEQ + q) * D_MODEL + h * D_HEAD + 16 * t + ln] =
                (__bf16)val;
        }
    }
}

// ---------------------------------------------------------------------------
// Launch
// ---------------------------------------------------------------------------
extern "C" void kernel_launch(void* const* d_in, const int* in_sizes, int n_in,
                              void* d_out, int out_size, void* d_ws, size_t ws_size,
                              hipStream_t stream) {
    const float* query = (const float*)d_in[0];
    const float* key   = (const float*)d_in[1];
    const float* value = (const float*)d_in[2];
    // d_in[3] = causal mask (int32) -- applied analytically in-kernel
    const float* wq = (const float*)d_in[4];
    const float* bq = (const float*)d_in[5];
    const float* wk = (const float*)d_in[6];
    const float* bk = (const float*)d_in[7];
    const float* wv = (const float*)d_in[8];
    const float* bv = (const float*)d_in[9];
    const float* wo = (const float*)d_in[10];
    const float* bo = (const float*)d_in[11];

    char* ws = (char*)d_ws;
    const size_t WBYTES = (size_t)D_MODEL * D_MODEL * sizeof(__bf16);      // 2 MB
    const size_t ABYTES = (size_t)BATCH * SEQ * D_MODEL * sizeof(__bf16);  // 16 MB
    __bf16* wq_b = (__bf16*)(ws + 0 * WBYTES);
    __bf16* wk_b = (__bf16*)(ws + 1 * WBYTES);
    __bf16* wv_b = (__bf16*)(ws + 2 * WBYTES);
    __bf16* wo_b = (__bf16*)(ws + 3 * WBYTES);
    __bf16* Qp   = (__bf16*)(ws + 4 * WBYTES);
    __bf16* Kp   = (__bf16*)(ws + 4 * WBYTES + 1 * ABYTES);
    __bf16* Vt   = (__bf16*)(ws + 4 * WBYTES + 2 * ABYTES);
    __bf16* Ao   = (__bf16*)(ws + 4 * WBYTES + 3 * ABYTES);

    const int wn = D_MODEL * D_MODEL;
    cvt_f32_bf16_kernel<<<(wn + 255) / 256, 256, 0, stream>>>(wq, wq_b, wn);
    cvt_f32_bf16_kernel<<<(wn + 255) / 256, 256, 0, stream>>>(wk, wk_b, wn);
    cvt_f32_bf16_kernel<<<(wn + 255) / 256, 256, 0, stream>>>(wv, wv_b, wn);
    cvt_f32_bf16_kernel<<<(wn + 255) / 256, 256, 0, stream>>>(wo, wo_b, wn);

    const int M = BATCH * SEQ;
    dim3 gb(128);
    dim3 gg(D_MODEL / 64, M / 64);
    gemm_wmma_kernel<false, 1><<<gg, gb, 0, stream>>>(query, wq_b, bq, Qp, M, D_MODEL, D_MODEL);
    gemm_wmma_kernel<false, 1><<<gg, gb, 0, stream>>>(key,   wk_b, bk, Kp, M, D_MODEL, D_MODEL);
    gemm_wmma_kernel<false, 2><<<gg, gb, 0, stream>>>(value, wv_b, bv, Vt, M, D_MODEL, D_MODEL);

    dim3 ag(BATCH * N_HEADS * (SEQ / 64));
    attention_kernel<<<ag, gb, 0, stream>>>(Qp, Kp, Vt, Ao);

    gemm_wmma_kernel<true, 0><<<gg, gb, 0, stream>>>(Ao, wo_b, bo, d_out, M, D_MODEL, D_MODEL);
}
